// ALSAP_30537217475289
// MI455X (gfx1250) — matrix operand order repaired and checked by reading, not compiled, not run on hardware
//
#include <hip/hip_runtime.h>
#include <cstdint>
#include <cstddef>

// ---------------- problem dims ----------------
static constexpr int kB = 16, kS = 128, kH = 512, kE = 128, kP = 64;
static constexpr int kNT = 32, kDB = 768, kWIN = 5;

typedef __bf16 bf16_t;
typedef __attribute__((ext_vector_type(16))) __bf16 v16bf;
typedef __attribute__((ext_vector_type(8)))  __bf16 v8bf;
typedef __attribute__((ext_vector_type(8)))  float  v8f;

// ---------------- scalar helpers ----------------
__device__ __forceinline__ float bf2f(bf16_t b) {
  unsigned short s = __builtin_bit_cast(unsigned short, b);
  unsigned u = ((unsigned)s) << 16;
  return __builtin_bit_cast(float, u);
}
__device__ __forceinline__ bf16_t f2bf(float f) {
  unsigned u = __builtin_bit_cast(unsigned, f);
  unsigned r = u + 0x7FFFu + ((u >> 16) & 1u);   // round-to-nearest-even
  unsigned short s = (unsigned short)(r >> 16);
  return __builtin_bit_cast(bf16_t, s);
}
__device__ __forceinline__ float sigf(float x) { return 1.f / (1.f + __expf(-x)); }

// ---------------- WMMA fragment helpers (CDNA5 16x16x32 bf16 layouts) ----------------
// A 16x32 (MxK), row-major, lane m=lane&15, contiguous K-halves per lane half (k = 16*(lane>>4) + s, this file's order, B matches)
__device__ __forceinline__ v16bf lda_frag(const bf16_t* __restrict__ A, int lda, int lane) {
  int m = lane & 15, hs = (lane >> 4) << 4;     // K sub-block: {hs..hs+15}
  const bf16_t* p0 = A + (size_t)m * lda + hs;
  v8bf a = *(const v8bf*)p0;
  v8bf b = *(const v8bf*)(p0 + 8);
  return __builtin_shufflevector(a, b, 0,1,2,3,4,5,6,7,8,9,10,11,12,13,14,15);
}
// B 32x16 (KxN) fed from weight matrix stored [N][K] row-major (PyTorch W): B[k][n]=W[n][k]
__device__ __forceinline__ v16bf ldbT_frag(const bf16_t* __restrict__ W, int ldw, int lane) {
  int n = lane & 15, ks = (lane >> 4) << 4;     // lane holds K = ks..ks+15 of column n
  const bf16_t* p = W + (size_t)n * ldw + ks;
  v8bf a = *(const v8bf*)p;
  v8bf b = *(const v8bf*)(p + 8);
  return __builtin_shufflevector(a, b, 0,1,2,3,4,5,6,7,8,9,10,11,12,13,14,15);
}
// B 32x16 (KxN) fed from an activation stored row-major [K][N] (strided gather)
__device__ __forceinline__ v16bf ldbN_frag(const bf16_t* __restrict__ Bm, int ldb, int lane) {
  int n = lane & 15, ks = (lane >> 4) << 4;
  v16bf f;
#pragma unroll
  for (int t = 0; t < 16; ++t) f[t] = Bm[(size_t)(ks + t) * ldb + n];
  return f;
}
__device__ __forceinline__ v8f wmma_bf16(v16bf a, v16bf b, v8f c) {
  return __builtin_amdgcn_wmma_f32_16x16x32_bf16(false, a, false, b, (short)0, c, false, false);
}
__device__ __forceinline__ v8f v8f_zero() {
  v8f z = {0.f,0.f,0.f,0.f,0.f,0.f,0.f,0.f};
  return z;
}
// in-wave softmax over 128 contiguous floats (4 per lane)
__device__ __forceinline__ void wave_softmax128(float* row, int lane) {
  float v[4]; float mx = -3.4e38f;
#pragma unroll
  for (int t = 0; t < 4; ++t) { v[t] = row[lane + 32 * t]; mx = fmaxf(mx, v[t]); }
#pragma unroll
  for (int o = 16; o; o >>= 1) mx = fmaxf(mx, __shfl_xor(mx, o, 32));
  float sum = 0.f;
#pragma unroll
  for (int t = 0; t < 4; ++t) { v[t] = __expf(v[t] - mx); sum += v[t]; }
#pragma unroll
  for (int o = 16; o; o >>= 1) sum += __shfl_xor(sum, o, 32);
  float inv = 1.f / sum;
#pragma unroll
  for (int t = 0; t < 4; ++t) row[lane + 32 * t] = v[t] * inv;
}

// ---------------- elementwise kernels ----------------
__global__ void k_cast_bf16(const float* __restrict__ src, bf16_t* __restrict__ dst, int n) {
  int i = blockIdx.x * blockDim.x + threadIdx.x;
  if (i < n) dst[i] = f2bf(src[i]);
}

__global__ void k_embed(const int* __restrict__ ids, const float* __restrict__ table,
                        bf16_t* __restrict__ out, int rows, int D, int special, int vocab) {
  int i = blockIdx.x * blockDim.x + threadIdx.x;
  if (i >= rows * D) return;
  int r = i / D, d = i - r * D;
  int v = ids[r];
  if (v < -1) v = special;
  if (v < 0) v += vocab;                 // python-style wrap
  out[i] = f2bf(table[(size_t)v * D + d]);
}

// subcat[b][s][w][h]: zero-padded sliding window over outputs
__global__ void k_subcat(const float* __restrict__ outputs, bf16_t* __restrict__ sub) {
  int i = blockIdx.x * blockDim.x + threadIdx.x;
  const int total = kB * kS * kWIN * kH;
  if (i >= total) return;
  int h = i % kH; int t = i / kH;
  int w = t % kWIN; t /= kWIN;
  int s = t % kS; int b = t / kS;
  int lo = s - 2 < 0 ? 0 : s - 2;
  int hi = s + 2 > kS - 1 ? kS - 1 : s + 2;
  float v = 0.f;
  if (w < hi - lo + 1) v = outputs[((size_t)b * kS + (lo + w)) * kH + h];
  sub[i] = f2bf(v);
}

// softmax over rows of 128; one wave per row; emit bf16
__global__ __launch_bounds__(256) void k_softmax128(const float* __restrict__ Sc,
                                                    bf16_t* __restrict__ Obf, int rows) {
  int wave = threadIdx.x >> 5, lane = threadIdx.x & 31;
  int row = blockIdx.x * 8 + wave;
  if (row >= rows) return;
  const float* p = Sc + (size_t)row * 128;
  float v[4]; float mx = -3.4e38f;
#pragma unroll
  for (int t = 0; t < 4; ++t) { v[t] = p[lane + 32 * t]; mx = fmaxf(mx, v[t]); }
#pragma unroll
  for (int o = 16; o; o >>= 1) mx = fmaxf(mx, __shfl_xor(mx, o, 32));
  float sum = 0.f;
#pragma unroll
  for (int t = 0; t < 4; ++t) { v[t] = __expf(v[t] - mx); sum += v[t]; }
#pragma unroll
  for (int o = 16; o; o >>= 1) sum += __shfl_xor(sum, o, 32);
  float inv = 1.f / sum;
#pragma unroll
  for (int t = 0; t < 4; ++t) Obf[(size_t)row * 128 + lane + 32 * t] = f2bf(v[t] * inv);
}

// ---------------- generic bf16 WMMA GEMM: C[M,N] = A[M,K] @ W[N,K]^T (+ bias) ----------------
__global__ __launch_bounds__(256) void k_gemm_bias(
    const bf16_t* __restrict__ A, const bf16_t* __restrict__ W,
    const float* __restrict__ bias, float* __restrict__ C, bf16_t* __restrict__ Cbf,
    int M, int N, int K) {
  const int lane = threadIdx.x & 31, wave = threadIdx.x >> 5;
  const int tiles_n = N >> 4;
  const int tile = blockIdx.x * 8 + wave;
  if (tile >= (M >> 4) * tiles_n) return;
  const int tm = tile / tiles_n, tn = tile - tm * tiles_n;
  const bf16_t* Ap = A + (size_t)tm * 16 * K;
  const bf16_t* Wp = W + (size_t)tn * 16 * K;
  v8f acc = v8f_zero();
  for (int k = 0; k < K; k += 32) {
    if (k + 512 < K) {                         // global_prefetch_b8 ahead on the deep-K GEMM
      __builtin_prefetch((const void*)(Ap + (size_t)(lane & 15) * K + k + 512), 0, 1);
    }
    acc = wmma_bf16(lda_frag(Ap + k, K, lane), ldbT_frag(Wp + k, K, lane), acc);
  }
  int n = lane & 15, hs = (lane >> 4) << 3;
  float bv = bias ? bias[tn * 16 + n] : 0.f;
#pragma unroll
  for (int r = 0; r < 8; ++r) {
    size_t idx = (size_t)(tm * 16 + hs + r) * N + tn * 16 + n;
    float v = acc[r] + bv;
    C[idx] = v;
    if (Cbf) Cbf[idx] = f2bf(v);
  }
}

// ---------------- batched attention scores: sc[b] = q[b] @ outputs[b]^T ----------------
__global__ __launch_bounds__(256) void k_attn_scores(const bf16_t* __restrict__ qbf,
                                                     const bf16_t* __restrict__ obf,
                                                     float* __restrict__ sc) {
  const int lane = threadIdx.x & 31, wave = threadIdx.x >> 5;
  const int b = blockIdx.x >> 3;
  const int tid = (blockIdx.x & 7) * 8 + wave;     // 0..63 tiles (8x8 of 16)
  const int tm = tid >> 3, tn = tid & 7;
  const bf16_t* A = qbf + ((size_t)b * kS + tm * 16) * kH;
  const bf16_t* W = obf + ((size_t)b * kS + tn * 16) * kH;   // W-form: rows j, contiguous K=H
  v8f acc = v8f_zero();
  for (int k = 0; k < kH; k += 32)
    acc = wmma_bf16(lda_frag(A + k, kH, lane), ldbT_frag(W + k, kH, lane), acc);
  int n = lane & 15, hs = (lane >> 4) << 3;
#pragma unroll
  for (int r = 0; r < 8; ++r)
    sc[((size_t)b * kS + tm * 16 + hs + r) * kS + tn * 16 + n] = acc[r];
}

// ---------------- nw[b] = attn[b] @ outputs[b]  (B operand row-major KxN) ----------------
__global__ __launch_bounds__(256) void k_attn_apply(const bf16_t* __restrict__ attnbf,
                                                    const bf16_t* __restrict__ obf,
                                                    float* __restrict__ nw) {
  const int lane = threadIdx.x & 31, wave = threadIdx.x >> 5;
  const int b = blockIdx.x >> 5;
  const int tid = (blockIdx.x & 31) * 8 + wave;    // 0..255 tiles (8 x 32 of 16)
  const int tm = tid >> 5, tn = tid & 31;
  const bf16_t* A = attnbf + ((size_t)b * kS + tm * 16) * kS;
  const bf16_t* Bm = obf + (size_t)b * kS * kH;
  v8f acc = v8f_zero();
  for (int k = 0; k < kS; k += 32)
    acc = wmma_bf16(lda_frag(A + k, kS, lane),
                    ldbN_frag(Bm + (size_t)k * kH + tn * 16, kH, lane), acc);
  int n = lane & 15, hs = (lane >> 4) << 3;
#pragma unroll
  for (int r = 0; r < 8; ++r)
    nw[((size_t)b * kS + tm * 16 + hs + r) * kH + tn * 16 + n] = acc[r];
}

// ---------------- pos LSTM recurrence (x-gates precomputed): block 0 = fwd, 1 = bwd ----------
__global__ __launch_bounds__(512) void k_pos_lstm(
    const float* __restrict__ gxf, const float* __restrict__ gxb,
    const bf16_t* __restrict__ Wfhh, const float* __restrict__ bfi, const float* __restrict__ bfh,
    const bf16_t* __restrict__ Wbhh, const float* __restrict__ bbi, const float* __restrict__ bbh,
    float* __restrict__ lstm_pos) {
  const int dir = blockIdx.x;
  const float*  gx  = dir ? gxb : gxf;
  const bf16_t* Whh = dir ? Wbhh : Wfhh;
  const float*  bi  = dir ? bbi : bfi;
  const float*  bh  = dir ? bbh : bfh;
  const int lane = threadIdx.x & 31, wave = threadIdx.x >> 5;   // 16 waves = 16 gate tiles
  __shared__ __align__(16) bf16_t hbuf[kB * kP];
  __shared__ float cbuf[kB * kP];
  __shared__ float gates[kB * 4 * kP];
  // hoist loop-invariant recurrent-weight fragments into registers
  v16bf wf[2];
#pragma unroll
  for (int k = 0; k < kP; k += 32)
    wf[k >> 5] = ldbT_frag(Whh + (size_t)wave * 16 * kP + k, kP, lane);
  for (int u = threadIdx.x; u < kB * kP; u += 512) { hbuf[u] = f2bf(0.f); cbuf[u] = 0.f; }
  __syncthreads();
  for (int st = 0; st < kS; ++st) {
    const int t = dir ? (kS - 1 - st) : st;
    v8f acc = v8f_zero();
#pragma unroll
    for (int k = 0; k < kP; k += 32)
      acc = wmma_bf16(lda_frag(hbuf + k, kP, lane), wf[k >> 5], acc);
    int n = lane & 15, hs = (lane >> 4) << 3;
#pragma unroll
    for (int r = 0; r < 8; ++r) gates[(hs + r) * 256 + wave * 16 + n] = acc[r];
    __syncthreads();
    for (int u = threadIdx.x; u < kB * kP; u += 512) {
      int b = u >> 6, e = u & 63;
      const float* g = gates + b * 256;
      const float* x = gx + ((size_t)b * kS + t) * 256;
      float gi = x[e]       + g[e]        + bi[e]        + bh[e];
      float gf = x[64 + e]  + g[64 + e]   + bi[64 + e]   + bh[64 + e];
      float gg = x[128 + e] + g[128 + e]  + bi[128 + e]  + bh[128 + e];
      float go = x[192 + e] + g[192 + e]  + bi[192 + e]  + bh[192 + e];
      float c = sigf(gf) * cbuf[u] + sigf(gi) * tanhf(gg);
      float h = sigf(go) * tanhf(c);
      cbuf[u] = c; hbuf[u] = f2bf(h);
      lstm_pos[((size_t)b * kS + t) * (2 * kP) + dir * kP + e] = h;
    }
    __syncthreads();
  }
}

// ---------------- forward y LSTM recurrence (x-gates precomputed) ----------------
__global__ __launch_bounds__(1024) void k_ylstm_fwd(
    const float* __restrict__ gx, const bf16_t* __restrict__ Whh,
    const float* __restrict__ bi, const float* __restrict__ bh, float* __restrict__ fy) {
  const int lane = threadIdx.x & 31, wave = threadIdx.x >> 5;   // 32 waves = 32 gate tiles
  __shared__ __align__(16) bf16_t hbuf[kB * kE];
  __shared__ float cbuf[kB * kE];
  __shared__ float gates[kB * 4 * kE];
  v16bf wf[4];
#pragma unroll
  for (int k = 0; k < kE; k += 32)
    wf[k >> 5] = ldbT_frag(Whh + (size_t)wave * 16 * kE + k, kE, lane);
  for (int u = threadIdx.x; u < kB * kE; u += 1024) { hbuf[u] = f2bf(0.f); cbuf[u] = 0.f; }
  __syncthreads();
  for (int t = 0; t < kS; ++t) {
    v8f acc = v8f_zero();
#pragma unroll
    for (int k = 0; k < kE; k += 32)
      acc = wmma_bf16(lda_frag(hbuf + k, kE, lane), wf[k >> 5], acc);
    int n = lane & 15, hs = (lane >> 4) << 3;
#pragma unroll
    for (int r = 0; r < 8; ++r) gates[(hs + r) * 512 + wave * 16 + n] = acc[r];
    __syncthreads();
    for (int u = threadIdx.x; u < kB * kE; u += 1024) {
      int b = u >> 7, e = u & 127;
      const float* g = gates + b * 512;
      const float* x = gx + ((size_t)b * kS + t) * 512;
      float gi = x[e]       + g[e]        + bi[e]        + bh[e];
      float gf = x[128 + e] + g[128 + e]  + bi[128 + e]  + bh[128 + e];
      float gg = x[256 + e] + g[256 + e]  + bi[256 + e]  + bh[256 + e];
      float go = x[384 + e] + g[384 + e]  + bi[384 + e]  + bh[384 + e];
      float c = sigf(gf) * cbuf[u] + sigf(gi) * tanhf(gg);
      float h = sigf(go) * tanhf(c);
      cbuf[u] = c; hbuf[u] = f2bf(h);
      fy[((size_t)b * kS + t) * kE + e] = h;
    }
    __syncthreads();
  }
}

// ---------------- timestep 0 prediction (se_linear_first branch) ----------------
__global__ __launch_bounds__(512) void k_pred0(
    const float* __restrict__ outputs, const float* __restrict__ nw,
    const float* __restrict__ lstm_pos,
    const float* __restrict__ W_tte, const float* __restrict__ b_tte,
    const float* __restrict__ W_se1, const float* __restrict__ b_se1,
    float* __restrict__ pred) {
  __shared__ float lbl0[kB * kE];
  for (int u = threadIdx.x; u < kB * kE; u += 512) {
    int b = u >> 7, e = u & 127;
    const float* w = W_tte + (size_t)e * (kH + 2 * kE);
    const float* nwp = nw + (size_t)b * kS * kH;      // position 0
    float s = b_tte[e];
    for (int h = 0; h < kH; ++h) s += nwp[h] * w[h];  // cols H..H+2E multiply zeros
    lbl0[u] = s;
  }
  __syncthreads();
  const int b = threadIdx.x >> 5, n = threadIdx.x & 31;
  const float* w = W_se1 + (size_t)n * 1920;
  const float* ob = outputs + (size_t)b * kS * kH;
  const float* nwp = nw + (size_t)b * kS * kH;
  const float* lp = lstm_pos + (size_t)b * kS * (2 * kP);
  float s = b_se1[n];
  for (int h = 0; h < kH; ++h) s += ob[h] * (w[h] + w[512 + h]);
  for (int h = 0; h < kH; ++h) s += nwp[h] * w[1024 + h];
  for (int e = 0; e < 2 * kP; ++e) s += lp[e] * w[1536 + e];
  for (int e = 0; e < kE; ++e) s += lbl0[b * kE + e] * (w[1664 + e] + w[1792 + e]);
  pred[(size_t)b * kS * kNT + n] = s;
}

// ---------------- per-position chain kernel: block i-1 handles timestep i ----------------
__global__ __launch_bounds__(1024) void k_chain(
    const float* __restrict__ gx, const bf16_t* __restrict__ Whh,
    const float* __restrict__ bi, const float* __restrict__ bh,
    const float* __restrict__ fy,
    const float* __restrict__ outputs, const float* __restrict__ nw,
    const float* __restrict__ lstm_pos,
    const float* __restrict__ W_lout, const float* __restrict__ b_lout,
    const float* __restrict__ W_b2t, const float* __restrict__ b_b2t,
    const float* __restrict__ W_se, const float* __restrict__ b_se,
    bf16_t* __restrict__ bwdstate, float* __restrict__ pred) {
  const int i = blockIdx.x + 1;                     // 1..127
  const int lane = threadIdx.x & 31, wave = threadIdx.x >> 5;
  __shared__ __align__(16) unsigned char smem[61440];
  bf16_t* hbuf  = (bf16_t*)smem;                    //  4 KB  [16][128] bf16
  float*  cbuf  = (float*)(smem + 4096);            //  8 KB  [16][128] (later: score/aw scratch)
  float*  gates = (float*)(smem + 12288);           // 32 KB  [16][512] (later label_mem -> label_bg)
  float*  relb  = (float*)(smem + 45056);           // 16 KB  [16][256] (later out_mem -> label_ctx)
  bf16_t* mystate = bwdstate + (size_t)i * kS * kB * kE;

  // hoist recurrent-weight fragments (reused up to 127x in the sequential loop)
  v16bf wf[4];
#pragma unroll
  for (int kk = 0; kk < kE; kk += 32)
    wf[kk >> 5] = ldbT_frag(Whh + (size_t)wave * 16 * kE + kk, kE, lane);
  for (int u = threadIdx.x; u < kB * kE; u += 1024) { hbuf[u] = f2bf(0.f); cbuf[u] = 0.f; }
  __syncthreads();

  // ---- Phase A: backward LSTM over prefix, exactly i steps (h-part via WMMA) ----
  for (int k = 0; k < i; ++k) {
    const int p = i - 1 - k;
    v8f acc = v8f_zero();
#pragma unroll
    for (int kk = 0; kk < kE; kk += 32)
      acc = wmma_bf16(lda_frag(hbuf + kk, kE, lane), wf[kk >> 5], acc);
    int n = lane & 15, hs = (lane >> 4) << 3;
#pragma unroll
    for (int r = 0; r < 8; ++r) gates[(hs + r) * 512 + wave * 16 + n] = acc[r];
    __syncthreads();
    for (int u = threadIdx.x; u < kB * kE; u += 1024) {
      int b = u >> 7, e = u & 127;
      const float* g = gates + b * 512;
      const float* x = gx + ((size_t)b * kS + p) * 512;
      float gi = x[e]       + g[e]        + bi[e]        + bh[e];
      float gf = x[128 + e] + g[128 + e]  + bi[128 + e]  + bh[128 + e];
      float gg = x[256 + e] + g[256 + e]  + bi[256 + e]  + bh[256 + e];
      float go = x[384 + e] + g[384 + e]  + bi[384 + e]  + bh[384 + e];
      float c = sigf(gf) * cbuf[u] + sigf(gi) * tanhf(gg);
      float h = sigf(go) * tanhf(c);
      cbuf[u] = c; hbuf[u] = f2bf(h);
      mystate[((size_t)p * kB + b) * kE + e] = f2bf(h);
      if (k == 0) {                                  // rel = [fwd_y[:,i-1], bwd at j=i-1]
        relb[b * 256 + 128 + e] = h;
        relb[b * 256 + e] = fy[((size_t)b * kS + (i - 1)) * kE + e];
      }
    }
    __syncthreads();
  }

  // ---- Phase B: label_mem = rel @ W_lout^T + b_lout  (VALU, into gates region) ----
  float* label_mem = gates;
  for (int u = threadIdx.x; u < kB * kH; u += 1024) {
    int b = u >> 9, n = u & 511;
    const float* w = W_lout + (size_t)n * 256;
    const float* r = relb + b * 256;
    float s = b_lout[n];
    for (int kk = 0; kk < 256; ++kk) s += r[kk] * w[kk];
    label_mem[u] = s;
  }
  __syncthreads();

  // ---- Phase C: aw = softmax_j(label_mem . outputs_j); label_bg = sum aw_j outputs_j ----
  float* s1 = cbuf;
  for (int u = threadIdx.x; u < kB * kS; u += 1024) {
    int b = u >> 7, j = u & 127;
    const float* lm = label_mem + b * 512;
    const float* ob = outputs + ((size_t)b * kS + j) * kH;
    float s = 0.f;
    for (int h = 0; h < kH; ++h) s += lm[h] * ob[h];
    s1[u] = s;
  }
  __syncthreads();
  if (wave < kB) wave_softmax128(s1 + wave * kS, lane);
  __syncthreads();
  float* label_bg = gates;                           // overwrite label_mem
  for (int u = threadIdx.x; u < kB * kH; u += 1024) {
    int b = u >> 9, h = u & 511;
    const float* aw = s1 + b * kS;
    float s = 0.f;
    for (int j = 0; j < kS; ++j) s += aw[j] * outputs[((size_t)b * kS + j) * kH + h];
    label_bg[u] = s;
  }
  __syncthreads();

  // ---- Phase D: out_mem = [nw_i, label_bg] @ W_b2t^T + b_b2t  (into relb region) ----
  float* out_mem = relb;
  for (int u = threadIdx.x; u < kB * 256; u += 1024) {
    int b = u >> 8, n = u & 255;
    const float* w = W_b2t + (size_t)n * 1024;
    const float* nwp = nw + ((size_t)b * kS + i) * kH;
    const float* lb = label_bg + b * 512;
    float s = b_b2t[n];
    for (int h = 0; h < kH; ++h) s += nwp[h] * w[h];
    for (int h = 0; h < kH; ++h) s += lb[h] * w[512 + h];
    out_mem[u] = s;
  }
  __syncthreads();

  // ---- Phase E: masked attention over out_lstm; label_ctx ----
  float* s2 = cbuf;
  for (int u = threadIdx.x; u < kB * kS; u += 1024) {
    int b = u >> 7, j = u & 127;
    float s = -3.4028235e38f;
    if (j < i) {
      s = 0.f;
      const float* om = out_mem + b * 256;
      const float* fj = fy + ((size_t)b * kS + j) * kE;
      for (int e = 0; e < kE; ++e) s += om[e] * fj[e];
      const bf16_t* bj = mystate + ((size_t)j * kB + b) * kE;
      for (int e = 0; e < kE; ++e) s += om[128 + e] * bf2f(bj[e]);
    }
    s2[u] = s;
  }
  __syncthreads();
  if (wave < kB) wave_softmax128(s2 + wave * kS, lane);
  __syncthreads();
  float* label_ctx = relb;                           // overwrite out_mem
  for (int u = threadIdx.x; u < kB * 256; u += 1024) {
    int b = u >> 8, e = u & 255;
    const float* aw = s2 + b * kS;
    float s = 0.f;
    if (e < kE) {
      for (int j = 0; j < i; ++j) s += aw[j] * fy[((size_t)b * kS + j) * kE + e];
    } else {
      for (int j = 0; j < i; ++j) s += aw[j] * bf2f(mystate[((size_t)j * kB + b) * kE + (e - 128)]);
    }
    label_ctx[u] = s;
  }
  __syncthreads();

  // ---- Phase F: pred[:, i, :] = re @ W_se^T + b_se ----
  if (threadIdx.x < kB * kNT) {
    int b = threadIdx.x >> 5, n = threadIdx.x & 31;
    const float* w = W_se + (size_t)n * 1920;
    const float* ob = outputs + ((size_t)b * kS + i) * kH;
    const float* nwp = nw + ((size_t)b * kS + i) * kH;
    const float* lp = lstm_pos + ((size_t)b * kS + i) * (2 * kP);
    const float* lb = label_bg + b * 512;
    const float* lc = label_ctx + b * 256;
    float s = b_se[n];
    for (int h = 0; h < kH; ++h) s += ob[h] * w[h];
    for (int h = 0; h < kH; ++h) s += nwp[h] * w[512 + h];
    for (int e = 0; e < 2 * kP; ++e) s += lp[e] * w[1024 + e];
    for (int h = 0; h < kH; ++h) s += lb[h] * w[1152 + h];
    for (int e = 0; e < 256; ++e) s += lc[e] * w[1664 + e];
    pred[((size_t)b * kS + i) * kNT + n] = s;
  }
}

// ---------------- host orchestration ----------------
extern "C" void kernel_launch(void* const* d_in, const int* in_sizes, int n_in,
                              void* d_out, int out_size, void* d_ws, size_t ws_size,
                              hipStream_t stream) {
  (void)in_sizes; (void)n_in; (void)out_size; (void)ws_size;
  const float* x_emb   = (const float*)d_in[0];
  const int*   y       = (const int*)d_in[1];
  const int*   pos_dat = (const int*)d_in[2];
  const float* W_word  = (const float*)d_in[3];
  const float* b_word  = (const float*)d_in[4];
  const float* pos_tab = (const float*)d_in[5];
  const float* tgt_tab = (const float*)d_in[6];
  const float* pWf_ih  = (const float*)d_in[7];
  const float* pWf_hh  = (const float*)d_in[8];
  const float* pbf_ih  = (const float*)d_in[9];
  const float* pbf_hh  = (const float*)d_in[10];
  const float* pWb_ih  = (const float*)d_in[11];
  const float* pWb_hh  = (const float*)d_in[12];
  const float* pbb_ih  = (const float*)d_in[13];
  const float* pbb_hh  = (const float*)d_in[14];
  const float* yWf_ih  = (const float*)d_in[15];
  const float* yWf_hh  = (const float*)d_in[16];
  const float* ybf_ih  = (const float*)d_in[17];
  const float* ybf_hh  = (const float*)d_in[18];
  const float* yWb_ih  = (const float*)d_in[19];
  const float* yWb_hh  = (const float*)d_in[20];
  const float* ybb_ih  = (const float*)d_in[21];
  const float* ybb_hh  = (const float*)d_in[22];
  const float* W_sub   = (const float*)d_in[23];
  const float* b_sub   = (const float*)d_in[24];
  const float* W_tte   = (const float*)d_in[25];
  const float* b_tte   = (const float*)d_in[26];
  const float* W_lout  = (const float*)d_in[27];
  const float* b_lout  = (const float*)d_in[28];
  const float* W_b2t   = (const float*)d_in[29];
  const float* b_b2t   = (const float*)d_in[30];
  const float* W_se    = (const float*)d_in[31];
  const float* b_se    = (const float*)d_in[32];
  const float* W_se1   = (const float*)d_in[33];
  const float* b_se1   = (const float*)d_in[34];
  float* pred = (float*)d_out;

  char* ws = (char*)d_ws;
  size_t off = 0;
  auto carve = [&](size_t bytes) -> char* {
    char* p = ws + off; off += (bytes + 255) & ~(size_t)255; return p;
  };
  bf16_t* xe_bf   = (bf16_t*)carve((size_t)kB * kS * kDB * 2);
  bf16_t* Ww_bf   = (bf16_t*)carve((size_t)kH * kDB * 2);
  bf16_t* Wsub_bf = (bf16_t*)carve((size_t)kH * (kWIN * kH) * 2);
  bf16_t* pWfi_bf = (bf16_t*)carve((size_t)4 * kP * kP * 2);
  bf16_t* pWfh_bf = (bf16_t*)carve((size_t)4 * kP * kP * 2);
  bf16_t* pWbi_bf = (bf16_t*)carve((size_t)4 * kP * kP * 2);
  bf16_t* pWbh_bf = (bf16_t*)carve((size_t)4 * kP * kP * 2);
  bf16_t* yWfi_bf = (bf16_t*)carve((size_t)4 * kE * kE * 2);
  bf16_t* yWfh_bf = (bf16_t*)carve((size_t)4 * kE * kE * 2);
  bf16_t* yWbi_bf = (bf16_t*)carve((size_t)4 * kE * kE * 2);
  bf16_t* yWbh_bf = (bf16_t*)carve((size_t)4 * kE * kE * 2);
  bf16_t* pe_bf   = (bf16_t*)carve((size_t)kB * kS * kP * 2);
  bf16_t* ye_bf   = (bf16_t*)carve((size_t)kB * kS * kE * 2);
  float*  out_f   = (float*)carve((size_t)kB * kS * kH * 4);
  bf16_t* out_bf  = (bf16_t*)carve((size_t)kB * kS * kH * 2);
  float*  lpos_f  = (float*)carve((size_t)kB * kS * 2 * kP * 4);
  float*  fy_f    = (float*)carve((size_t)kB * kS * kE * 4);
  bf16_t* sub_bf  = (bf16_t*)carve((size_t)kB * kS * kWIN * kH * 2);
  float*  q_f     = (float*)carve((size_t)kB * kS * kH * 4);
  bf16_t* q_bf    = (bf16_t*)carve((size_t)kB * kS * kH * 2);
  float*  sc_f    = (float*)carve((size_t)kB * kS * kS * 4);
  bf16_t* attn_bf = (bf16_t*)carve((size_t)kB * kS * kS * 2);
  float*  nw_f    = (float*)carve((size_t)kB * kS * kH * 4);
  float*  gpf_f   = (float*)carve((size_t)kB * kS * 4 * kP * 4);   // pos fwd x-gates
  float*  gpb_f   = (float*)carve((size_t)kB * kS * 4 * kP * 4);   // pos bwd x-gates
  float*  gyf_f   = (float*)carve((size_t)kB * kS * 4 * kE * 4);   // y fwd x-gates
  float*  gyb_f   = (float*)carve((size_t)kB * kS * 4 * kE * 4);   // y bwd x-gates (shared by all chains)
  bf16_t* bwd_bf  = (bf16_t*)carve((size_t)kS * kS * kB * kE * 2); // [chain i][j][b][e]

  auto cast = [&](const float* src, bf16_t* dst, int n) {
    k_cast_bf16<<<(n + 255) / 256, 256, 0, stream>>>(src, dst, n);
  };
  cast(x_emb,  xe_bf,   kB * kS * kDB);
  cast(W_word, Ww_bf,   kH * kDB);
  cast(W_sub,  Wsub_bf, kH * kWIN * kH);
  cast(pWf_ih, pWfi_bf, 4 * kP * kP);
  cast(pWf_hh, pWfh_bf, 4 * kP * kP);
  cast(pWb_ih, pWbi_bf, 4 * kP * kP);
  cast(pWb_hh, pWbh_bf, 4 * kP * kP);
  cast(yWf_ih, yWfi_bf, 4 * kE * kE);
  cast(yWf_hh, yWfh_bf, 4 * kE * kE);
  cast(yWb_ih, yWbi_bf, 4 * kE * kE);
  cast(yWb_hh, yWbh_bf, 4 * kE * kE);

  // embeddings
  k_embed<<<(kB * kS * kP + 255) / 256, 256, 0, stream>>>(
      pos_dat, pos_tab, pe_bf, kB * kS, kP, /*special=*/51, /*vocab=*/52);
  k_embed<<<(kB * kS * kE + 255) / 256, 256, 0, stream>>>(
      y, tgt_tab, ye_bf, kB * kS, kE, /*special=*/33, /*vocab=*/34);

  // outputs = x_emb @ W_word^T + b_word   (M=2048,N=512,K=768)
  k_gemm_bias<<<(kB * kS / 16) * (kH / 16) / 8, 256, 0, stream>>>(
      xe_bf, Ww_bf, b_word, out_f, out_bf, kB * kS, kH, kDB);

  // precompute sequence-parallel x-gates for all four LSTMs (dense WMMA GEMMs)
  k_gemm_bias<<<(kB * kS / 16) * (4 * kP / 16) / 8, 256, 0, stream>>>(
      pe_bf, pWfi_bf, nullptr, gpf_f, nullptr, kB * kS, 4 * kP, kP);
  k_gemm_bias<<<(kB * kS / 16) * (4 * kP / 16) / 8, 256, 0, stream>>>(
      pe_bf, pWbi_bf, nullptr, gpb_f, nullptr, kB * kS, 4 * kP, kP);
  k_gemm_bias<<<(kB * kS / 16) * (4 * kE / 16) / 8, 256, 0, stream>>>(
      ye_bf, yWfi_bf, nullptr, gyf_f, nullptr, kB * kS, 4 * kE, kE);
  k_gemm_bias<<<(kB * kS / 16) * (4 * kE / 16) / 8, 256, 0, stream>>>(
      ye_bf, yWbi_bf, nullptr, gyb_f, nullptr, kB * kS, 4 * kE, kE);

  // LSTM recurrences (h-part only)
  k_pos_lstm<<<2, 512, 0, stream>>>(gpf_f, gpb_f, pWfh_bf, pbf_ih, pbf_hh,
                                    pWbh_bf, pbb_ih, pbb_hh, lpos_f);
  k_ylstm_fwd<<<1, 1024, 0, stream>>>(gyf_f, yWfh_bf, ybf_ih, ybf_hh, fy_f);

  // sliding window features -> q
  k_subcat<<<(kB * kS * kWIN * kH + 255) / 256, 256, 0, stream>>>(out_f, sub_bf);
  k_gemm_bias<<<(kB * kS / 16) * (kH / 16) / 8, 256, 0, stream>>>(
      sub_bf, Wsub_bf, b_sub, q_f, q_bf, kB * kS, kH, kWIN * kH);

  // self-attention -> nw
  k_attn_scores<<<kB * 8, 256, 0, stream>>>(q_bf, out_bf, sc_f);
  k_softmax128<<<(kB * kS + 7) / 8, 256, 0, stream>>>(sc_f, attn_bf, kB * kS);
  k_attn_apply<<<kB * 32, 256, 0, stream>>>(attn_bf, out_bf, nw_f);

  // predictions
  k_pred0<<<1, 512, 0, stream>>>(out_f, nw_f, lpos_f, W_tte, b_tte, W_se1, b_se1, pred);
  k_chain<<<kS - 1, 1024, 0, stream>>>(
      gyb_f, yWbh_bf, ybb_ih, ybb_hh, fy_f, out_f, nw_f, lpos_f,
      W_lout, b_lout, W_b2t, b_b2t, W_se, b_se, bwd_bf, pred);
}
